// SelfAttention_26920855011564
// MI455X (gfx1250) — matrix operand order repaired and checked by reading, not compiled, not run on hardware
//
#include <hip/hip_runtime.h>

typedef __attribute__((ext_vector_type(16))) __bf16 v16bf;
typedef __attribute__((ext_vector_type(8)))  __bf16 v8bf;
typedef __attribute__((ext_vector_type(8)))  float  v8f;

static __device__ inline v8f wmma_bf16(v16bf a, v16bf b, v8f c) {
  return __builtin_amdgcn_wmma_f32_16x16x32_bf16(false, a, false, b, (short)0, c,
                                                 false, false);
}

// A fragment: 16x32 (MxK) bf16, row-major source (row stride ld elems).
// Contiguous k order (B uses the same): lane m=L&15, half h=L>>4 holds K in [16h,16h+16).
static __device__ inline v16bf load_a_bf16(const __bf16* base, int ld) {
  const int lane = threadIdx.x & 31;
  const int m = lane & 15, h = lane >> 4;
  const __bf16* p = base + (size_t)m * ld + h * 16;
  v16bf f;
  *((v8bf*)&f)     = *((const v8bf*)p);
  *((v8bf*)&f + 1) = *((const v8bf*)(p + 8));
  return f;
}

// B fragment: 32x16 (KxN) bf16 from B^T row-major (N rows x K cols), 32B-aligned rows.
// Contiguous k order (matches A): lane n=L&15, half h holds K in [16h,16h+16) -> one contiguous 32B.
static __device__ inline v16bf load_b_bf16(const __bf16* baseT, int ld) {
  const int lane = threadIdx.x & 31;
  const int nn = lane & 15, h = lane >> 4;
  return *((const v16bf*)(baseT + (size_t)nn * ld + h * 16));
}

// Same B fragment via two 16B loads (for padded LDS tiles, 16B alignment only).
static __device__ inline v16bf load_b_16al(const __bf16* baseT, int ld) {
  const int lane = threadIdx.x & 31;
  const int nn = lane & 15, h = lane >> 4;
  const __bf16* p = baseT + (size_t)nn * ld + h * 16;
  v16bf f;
  *((v8bf*)&f)     = *((const v8bf*)p);
  *((v8bf*)&f + 1) = *((const v8bf*)(p + 8));
  return f;
}

// Async global->LDS 16B per lane (GLOBAL_LOAD_ASYNC_TO_LDS_B128, ASYNCcnt).
#define ASYNC_G2L_B128(ldsoff, gptr, IMM)                                     \
  asm volatile("global_load_async_to_lds_b128 %0, %1, off offset:" IMM        \
               :                                                              \
               : "v"(ldsoff), "v"(gptr)                                       \
               : "memory")

// ---------------- weights fp32 -> bf16 (one shot) ----------------
__global__ __launch_bounds__(256) void wcvt_kernel(const float* __restrict__ qw,
                                                   const float* __restrict__ pw,
                                                   __bf16* __restrict__ wq,
                                                   __bf16* __restrict__ wp) {
  const int i = blockIdx.x * 256 + threadIdx.x;
  if (i < 768 * 256) wq[i] = (__bf16)qw[i];
  if (i < 256 * 256) wp[i] = (__bf16)pw[i];
}

// ---------------- GroupNorm -> hT (B,N,C) bf16 ----------------
__global__ __launch_bounds__(256) void gn_kernel(const float* __restrict__ x,
                                                 const float* __restrict__ w,
                                                 const float* __restrict__ bia,
                                                 __bf16* __restrict__ hT) {
  const int b = blockIdx.x >> 5;
  const int g = blockIdx.x & 31;
  const int t = threadIdx.x;
  const float* xg = x + ((size_t)b * 256 + g * 8) * 4096;
  float s = 0.f, s2 = 0.f;
  for (int u = t; u < 32768; u += 256) {
    float vv = xg[u];
    s += vv;
    s2 += vv * vv;
  }
  __shared__ float rs[256], rs2[256];
  rs[t] = s; rs2[t] = s2;
  __syncthreads();
  for (int off = 128; off > 0; off >>= 1) {
    if (t < off) { rs[t] += rs[t + off]; rs2[t] += rs2[t + off]; }
    __syncthreads();
  }
  const float mean = rs[0] * (1.f / 32768.f);
  const float var  = rs2[0] * (1.f / 32768.f) - mean * mean;
  const float rstd = rsqrtf(var + 1e-5f);
  for (int u = t; u < 32768; u += 256) {
    const int cl = u >> 12, nn = u & 4095;
    const int c = g * 8 + cl;
    const float vv = (xg[u] - mean) * rstd * w[c] + bia[c];
    hT[((size_t)b * 4096 + nn) * 256 + c] = (__bf16)vv;
  }
}

// ---------------- QKV GEMM: (768x256)@(256x4096) per batch ----------------
__global__ __launch_bounds__(256) void qkv_kernel(const __bf16* __restrict__ W,
                                                  const float* __restrict__ bias,
                                                  const __bf16* __restrict__ hT,
                                                  __bf16* __restrict__ qT,
                                                  __bf16* __restrict__ kT,
                                                  __bf16* __restrict__ vv) {
  const int b = blockIdx.z;
  const int wid = threadIdx.x >> 5;
  const int mw = wid >> 1, nw = wid & 1;
  const int o_base = blockIdx.y * 128 + mw * 32;
  const int n_base = blockIdx.x * 64 + nw * 32;
  const __bf16* hTb = hT + (size_t)b * 4096 * 256;
  v8f acc[2][2] = {};
#pragma unroll
  for (int kt = 0; kt < 8; ++kt) {
    v16bf a0 = load_a_bf16(W + (size_t)o_base * 256 + kt * 32, 256);
    v16bf a1 = load_a_bf16(W + (size_t)(o_base + 16) * 256 + kt * 32, 256);
    v16bf b0 = load_b_bf16(hTb + (size_t)n_base * 256 + kt * 32, 256);
    v16bf b1 = load_b_bf16(hTb + (size_t)(n_base + 16) * 256 + kt * 32, 256);
    acc[0][0] = wmma_bf16(a0, b0, acc[0][0]);
    acc[0][1] = wmma_bf16(a0, b1, acc[0][1]);
    acc[1][0] = wmma_bf16(a1, b0, acc[1][0]);
    acc[1][1] = wmma_bf16(a1, b1, acc[1][1]);
  }
  const int lane = threadIdx.x & 31;
  const int nn = lane & 15, h = lane >> 4;
#pragma unroll
  for (int tm = 0; tm < 2; ++tm) {
#pragma unroll
    for (int tn = 0; tn < 2; ++tn) {
      const int o0 = o_base + tm * 16;
      const int n0 = n_base + tn * 16;
      if (o0 < 256) {  // q -> qT (N,C), softmax scale folded in
        v8bf st;
#pragma unroll
        for (int r = 0; r < 8; ++r)
          st[r] = (__bf16)((acc[tm][tn][r] + bias[o0 + 8 * h + r]) * 0.0625f);
        *(v8bf*)(qT + ((size_t)b * 4096 + n0 + nn) * 256 + o0 + 8 * h) = st;
      } else if (o0 < 512) {  // k -> kT (N,C)
        v8bf st;
#pragma unroll
        for (int r = 0; r < 8; ++r)
          st[r] = (__bf16)(acc[tm][tn][r] + bias[o0 + 8 * h + r]);
        *(v8bf*)(kT + ((size_t)b * 4096 + n0 + nn) * 256 + (o0 - 256) + 8 * h) = st;
      } else {  // v -> (C,N); per-register stores are lane-contiguous across n
#pragma unroll
        for (int r = 0; r < 8; ++r)
          vv[((size_t)b * 256 + (o0 - 512) + 8 * h + r) * 4096 + n0 + nn] =
              (__bf16)(acc[tm][tn][r] + bias[o0 + 8 * h + r]);
      }
    }
  }
}

// ---------------- Flash attention ----------------
// 8 waves/block; each wave owns 16 query rows. V (256x32 per 32-j chunk) is
// staged into double-buffered padded LDS via async global->LDS loads (one chunk
// of lookahead hides the latency). K streams from global: row-linear, shared by
// all 8 waves within an iteration, so the WGP$ serves it.
#define VLD 40  // 256x40 bf16 rows -> conflict-free fragment reads
#define PLD 40  // per-wave P tile 16x40
__global__ __launch_bounds__(256) void attn_kernel(const __bf16* __restrict__ qT,
                                                   const __bf16* __restrict__ kT,
                                                   const __bf16* __restrict__ vv,
                                                   __bf16* __restrict__ aoT) {
  __shared__ alignas(32) __bf16 Vt[2][256 * VLD];
  __shared__ alignas(32) __bf16 Pb[8][16 * PLD];
  const int t = threadIdx.x;
  const int lane = t & 31;
  const int wid = t >> 5;
  const int b = blockIdx.y;
  const int i0 = blockIdx.x * 128 + wid * 16;
  const int nn = lane & 15, h = lane >> 4;
  const __bf16* qTb = qT + (size_t)b * 4096 * 256;
  const __bf16* kTb = kT + (size_t)b * 4096 * 256;
  const __bf16* vb  = vv + (size_t)b * 256 * 4096;
  // V staging: thread t copies channel row c=t, 64B per chunk
  const __bf16* gvbase = vb + (size_t)t * 4096;
  const unsigned lv0 = (unsigned)(unsigned long long)(Vt[0] + t * VLD);
  const unsigned lv1 = (unsigned)(unsigned long long)(Vt[1] + t * VLD);
  v16bf qa[8];
#pragma unroll
  for (int ct = 0; ct < 8; ++ct)
    qa[ct] = load_a_bf16(qTb + (size_t)i0 * 256 + ct * 32, 256);
  v8f O[16] = {};
  float Mr[8], Lr[8];
#pragma unroll
  for (int r = 0; r < 8; ++r) { Mr[r] = -1e30f; Lr[r] = 0.f; }
  __bf16* P = Pb[wid];
  {  // prologue: chunk 0 -> buffer 0
    const __bf16* gv = gvbase;
    ASYNC_G2L_B128(lv0, gv, "0");
    ASYNC_G2L_B128(lv0, gv, "16");
    ASYNC_G2L_B128(lv0, gv, "32");
    ASYNC_G2L_B128(lv0, gv, "48");
  }
  for (int jc = 0, it = 0; jc < 4096; jc += 32, ++it) {
    if (jc + 32 < 4096) {  // prefetch chunk it+1 into the other buffer
      const __bf16* gv = gvbase + jc + 32;
      const unsigned lv = (it & 1) ? lv0 : lv1;
      ASYNC_G2L_B128(lv, gv, "0");
      ASYNC_G2L_B128(lv, gv, "16");
      ASYNC_G2L_B128(lv, gv, "32");
      ASYNC_G2L_B128(lv, gv, "48");
      // async ops complete in order: cnt<=4 means chunk `it` has landed
      asm volatile("s_wait_asynccnt 0x4" ::: "memory");
    } else {
      asm volatile("s_wait_asynccnt 0x0" ::: "memory");
    }
    __syncthreads();  // all waves' deposits for chunk `it` visible
    const __bf16* Vcur = Vt[it & 1];
    v8f S0 = {}, S1 = {};
#pragma unroll
    for (int ct = 0; ct < 8; ++ct) {  // S = (q*scale)^T k over C=256, K from L2/WGP$
      v16bf kb0 = load_b_bf16(kTb + (size_t)jc * 256 + ct * 32, 256);
      v16bf kb1 = load_b_bf16(kTb + (size_t)(jc + 16) * 256 + ct * 32, 256);
      S0 = wmma_bf16(qa[ct], kb0, S0);
      S1 = wmma_bf16(qa[ct], kb1, S1);
    }
    // Online softmax; row i = r + 8h lives in one 16-lane half -> xor butterflies.
    float alpha[8];
#pragma unroll
    for (int r = 0; r < 8; ++r) {
      float mx = fmaxf(S0[r], S1[r]);
      mx = fmaxf(mx, __shfl_xor(mx, 1));
      mx = fmaxf(mx, __shfl_xor(mx, 2));
      mx = fmaxf(mx, __shfl_xor(mx, 4));
      mx = fmaxf(mx, __shfl_xor(mx, 8));
      const float newM = fmaxf(Mr[r], mx);
      alpha[r] = __expf(Mr[r] - newM);
      const float p0 = __expf(S0[r] - newM);
      const float p1 = __expf(S1[r] - newM);
      float ps = p0 + p1;
      ps += __shfl_xor(ps, 1);
      ps += __shfl_xor(ps, 2);
      ps += __shfl_xor(ps, 4);
      ps += __shfl_xor(ps, 8);
      Lr[r] = Lr[r] * alpha[r] + ps;
      Mr[r] = newM;
      P[(r + 8 * h) * PLD + nn] = (__bf16)p0;  // D-layout -> LDS (i,j)
      P[(r + 8 * h) * PLD + 16 + nn] = (__bf16)p1;
    }
#pragma unroll
    for (int tt = 0; tt < 16; ++tt)
#pragma unroll
      for (int r = 0; r < 8; ++r) O[tt][r] *= alpha[r];
    // reload P in A-fragment layout (per-wave region; same-wave DS is in-order)
    v16bf pa;
    {
      const __bf16* p = P + (size_t)(lane & 15) * PLD + h * 16;
      *((v8bf*)&pa)     = *((const v8bf*)p);
      *((v8bf*)&pa + 1) = *((const v8bf*)(p + 8));
    }
#pragma unroll
    for (int ct2 = 0; ct2 < 16; ++ct2) {  // O += P @ V^T over the 32-j chunk
      v16bf bvf = load_b_16al(Vcur + (size_t)(ct2 * 16) * VLD, VLD);
      O[ct2] = wmma_bf16(pa, bvf, O[ct2]);
    }
    __syncthreads();  // chunk `it` consumed; its buffer may be refilled next iter
  }
  float inv[8];
#pragma unroll
  for (int r = 0; r < 8; ++r) inv[r] = 1.f / Lr[r];
#pragma unroll
  for (int ct2 = 0; ct2 < 16; ++ct2)
#pragma unroll
    for (int r = 0; r < 8; ++r)
      aoT[((size_t)b * 4096 + i0 + r + 8 * h) * 256 + ct2 * 16 + nn] =
          (__bf16)(O[ct2][r] * inv[r]);
}

// ---------------- Proj GEMM + bias + residual (fp32 out) ----------------
__global__ __launch_bounds__(256) void proj_kernel(const __bf16* __restrict__ W,
                                                   const float* __restrict__ bias,
                                                   const __bf16* __restrict__ aoT,
                                                   const float* __restrict__ x,
                                                   float* __restrict__ out) {
  const int b = blockIdx.z;
  const int wid = threadIdx.x >> 5;
  const int mw = wid & 1, nw = wid >> 1;
  const int o_base = blockIdx.y * 64 + mw * 32;
  const int n_base = blockIdx.x * 128 + nw * 32;
  const __bf16* aoTb = aoT + (size_t)b * 4096 * 256;
  v8f acc[2][2] = {};
#pragma unroll
  for (int kt = 0; kt < 8; ++kt) {
    v16bf a0 = load_a_bf16(W + (size_t)o_base * 256 + kt * 32, 256);
    v16bf a1 = load_a_bf16(W + (size_t)(o_base + 16) * 256 + kt * 32, 256);
    v16bf b0 = load_b_bf16(aoTb + (size_t)n_base * 256 + kt * 32, 256);
    v16bf b1 = load_b_bf16(aoTb + (size_t)(n_base + 16) * 256 + kt * 32, 256);
    acc[0][0] = wmma_bf16(a0, b0, acc[0][0]);
    acc[0][1] = wmma_bf16(a0, b1, acc[0][1]);
    acc[1][0] = wmma_bf16(a1, b0, acc[1][0]);
    acc[1][1] = wmma_bf16(a1, b1, acc[1][1]);
  }
  const int lane = threadIdx.x & 31;
  const int nn = lane & 15, h = lane >> 4;
#pragma unroll
  for (int tm = 0; tm < 2; ++tm)
#pragma unroll
    for (int tn = 0; tn < 2; ++tn) {
      const int o0 = o_base + tm * 16;
      const int n0 = n_base + tn * 16;
#pragma unroll
      for (int r = 0; r < 8; ++r) {
        const size_t idx = ((size_t)b * 256 + o0 + 8 * h + r) * 4096 + n0 + nn;
        out[idx] = x[idx] + acc[tm][tn][r] + bias[o0 + 8 * h + r];
      }
    }
}

extern "C" void kernel_launch(void* const* d_in, const int* in_sizes, int n_in,
                              void* d_out, int out_size, void* d_ws, size_t ws_size,
                              hipStream_t stream) {
  const float* x      = (const float*)d_in[0];
  const float* gn_w   = (const float*)d_in[1];
  const float* gn_b   = (const float*)d_in[2];
  const float* qkv_w  = (const float*)d_in[3];
  const float* qkv_b  = (const float*)d_in[4];
  const float* proj_w = (const float*)d_in[5];
  const float* proj_b = (const float*)d_in[6];
  float* out = (float*)d_out;

  const size_t M4 = (size_t)4 * 4096 * 256;  // elems per (B,N,C) bf16 tensor
  __bf16* hT  = (__bf16*)d_ws;
  __bf16* qT  = hT + M4;
  __bf16* kT  = qT + M4;
  __bf16* vv  = kT + M4;
  __bf16* aoT = vv + M4;
  __bf16* wq  = aoT + M4;            // 768x256 bf16
  __bf16* wp  = wq + 768 * 256;      // 256x256 bf16  (~40.5 MB total)

  wcvt_kernel<<<dim3(768), dim3(256), 0, stream>>>(qkv_w, proj_w, wq, wp);
  gn_kernel<<<dim3(128), dim3(256), 0, stream>>>(x, gn_w, gn_b, hT);
  qkv_kernel<<<dim3(64, 6, 4), dim3(256), 0, stream>>>(wq, qkv_b, hT, qT, kT, vv);
  attn_kernel<<<dim3(32, 4), dim3(256), 0, stream>>>(qT, kT, vv, aoT);
  proj_kernel<<<dim3(32, 4, 4), dim3(256), 0, stream>>>(wp, proj_b, aoT, x, out);
}
